// Foo_6390911337089
// MI455X (gfx1250) — matrix operand order repaired and checked by reading, not compiled, not run on hardware
//
#include <hip/hip_runtime.h>
#include <hip/hip_bf16.h>

typedef __attribute__((ext_vector_type(16))) __bf16 v16bf;
typedef __attribute__((ext_vector_type(8)))  __bf16 v8bf;
typedef __attribute__((ext_vector_type(8)))  float  v8f;
typedef __attribute__((ext_vector_type(4)))  int    v4i;

typedef __attribute__((address_space(1))) v4i* gv4i_t;   // global
typedef __attribute__((address_space(3))) v4i* lv4i_t;   // LDS

#define NROWS   65536
#define D_IN    3072
#define H       100
#define HP      112          // H padded to 7*16 (N tiles)
#define KP      128          // hidden K padded to 4*32
#define NT      7            // HP/16
#define OUT_N   10
#define CUTOFF  0.5f
#define WAVES   8
#define THREADS 256
#define ROWS_PER_BLOCK (WAVES * 16)

#define KCH     64           // stage-0 K chunk staged in LDS
#define NCH     (D_IN / KCH) // 48 chunks
#define NPACK   128          // stage-0 packed rows (112 used, pad for exact xfers)

// LDS row strides padded for conflict-free ds_load_b128 (64 banks x 4B)
#define KLA     72           // wbufA row stride: 144B = 36 dword-banks (9 mod 16 odd)
#define KLH     136          // wbufH/hbuf row stride: 272B = 68 dword-banks

// workspace element counts (bf16), global layouts stay contiguous
#define WA_ELEMS (NCH * NPACK * KCH)  // 393216  (chunk-contiguous)
#define WH_ELEMS (14 * HP * KP)       // 200704  (layer-contiguous, 28672B each)
#define WO_ELEMS (3 * 16 * KP)        // 6144

// ---------- CDNA5 async global->LDS staging (ASYNCcnt) with safe fallback ----------
#if __has_builtin(__builtin_amdgcn_global_load_async_to_lds_b128) && \
    __has_builtin(__builtin_amdgcn_s_wait_asynccnt)
#define HAVE_ASYNC_LDS 1
#else
#define HAVE_ASYNC_LDS 0
#endif

static __device__ __forceinline__ void stage_b128(const __bf16* g, __bf16* l) {
#if HAVE_ASYNC_LDS
  __builtin_amdgcn_global_load_async_to_lds_b128(
      (gv4i_t)(v4i*)const_cast<__bf16*>(g),
      (lv4i_t)(v4i*)l, 0, 0);
#else
  *(float4*)l = *(const float4*)g;   // sync: global_load_b128 + ds_store_b128
#endif
}

#if HAVE_ASYNC_LDS
#define WAIT_ASYNC(n) __builtin_amdgcn_s_wait_asynccnt(n)
#else
#define WAIT_ASYNC(n) ((void)0)      // sync path: DScnt ordering + barrier suffices
#endif

// ---------------- helpers ----------------

static __device__ __forceinline__ v8f wmma_bf16(v16bf a, v16bf b, v8f c) {
  return __builtin_amdgcn_wmma_f32_16x16x32_bf16(false, a, false, b, (short)0, c,
                                                 false, false);
}

static __device__ __forceinline__ v16bf cat16(v8bf lo, v8bf hi) {
  union { v16bf v; v8bf h[2]; } u;
  u.h[0] = lo; u.h[1] = hi;
  return u.v;
}

// B fragment: bf16 at K, K+16 (two 16B LDS loads; rows are only 16B-aligned)
static __device__ __forceinline__ v16bf ldB(const __bf16* p) {
  return cat16(*(const v8bf*)p, *(const v8bf*)(p + 16));
}

static __device__ __forceinline__ v8bf cvt8(float4 a, float4 b) {
  v8bf r;
  r[0] = (__bf16)a.x; r[1] = (__bf16)a.y; r[2] = (__bf16)a.z; r[3] = (__bf16)a.w;
  r[4] = (__bf16)b.x; r[5] = (__bf16)b.y; r[6] = (__bf16)b.z; r[7] = (__bf16)b.w;
  return r;
}

// ---------------- Weight prep: transpose + pad + fp32->bf16 ----------------

// stage-0 weights packed chunk-contiguous: wA[c][n][kk], n in [0,128), kk in [0,64)
__global__ void prep_wa(const float* __restrict__ w_in, __bf16* __restrict__ wA) {
  int i = blockIdx.x * 256 + threadIdx.x;
  if (i >= WA_ELEMS) return;
  int c  = i / (NPACK * KCH);
  int r  = i % (NPACK * KCH);
  int n  = r / KCH;
  int kk = r % KCH;
  int k  = c * KCH + kk;
  float v = (n < H) ? w_in[(size_t)k * H + n] : 0.0f;
  wA[i] = (__bf16)v;
}

__global__ void prep_wh(const float* __restrict__ w_hid, __bf16* __restrict__ wH) {
  int i = blockIdx.x * 256 + threadIdx.x;
  if (i >= WH_ELEMS) return;
  int l = i / (HP * KP);
  int r = i % (HP * KP);
  int n = r / KP, k = r % KP;
  float v = (n < H && k < H) ? w_hid[((size_t)l * H + k) * H + n] : 0.0f;
  wH[i] = (__bf16)v;
}

__global__ void prep_wo(const float* __restrict__ w_out, __bf16* __restrict__ wO) {
  int i = blockIdx.x * 256 + threadIdx.x;
  if (i >= WO_ELEMS) return;
  int s = i / (16 * KP);
  int r = i % (16 * KP);
  int n = r / KP, k = r % KP;
  float v = (n < OUT_N && k < H) ? w_out[((size_t)s * H + k) * OUT_N + n] : 0.0f;
  wO[i] = (__bf16)v;
}

// ---------------- Fused 16-layer MLP cascade ----------------

__global__ __launch_bounds__(THREADS) void mlp_cascade_fused(
    const float* __restrict__ x,
    const float* __restrict__ b_in,
    const float* __restrict__ b_hid,
    const float* __restrict__ b_out,
    const __bf16* __restrict__ wA,
    const __bf16* __restrict__ wH,
    const __bf16* __restrict__ wO,
    float* __restrict__ out) {
  __shared__ __align__(16) __bf16 hbuf[WAVES][16][KLH];     // 34 KB h handoff
  __shared__ __align__(16) float  obuf[WAVES][3][16][16];   // 24 KB head outputs
  __shared__ __align__(16) __bf16 wbufA[2][NPACK][KLA];     // 36 KB stage-0 W tiles
  __shared__ __align__(16) __bf16 wbufH[2][HP][KLH];        // 60 KB hidden W tiles

  const int tid    = threadIdx.x;
  const int wave   = tid >> 5;
  const int lane   = tid & 31;
  const int laneLo = lane & 15;
  const int laneHi = lane >> 4;
  const int rowBase = blockIdx.x * ROWS_PER_BLOCK + wave * 16;
  const int kOffA  = laneHi ? 8  : 0;   // A-fragment K sub-offset
  const int kOffB  = laneHi ? 8  : 0;   // B-fragment K sub-offset

  // stage one stage-0 weight chunk: 1024 b128 xfers = 4/thread exact,
  // contiguous in global, scattered into padded LDS rows (8 xfers per 64-elem row)
  auto issue_wA = [&](int c) {
    const __bf16* src = wA + (size_t)c * (NPACK * KCH);
    __bf16* dstbase = &wbufA[c & 1][0][0];
#pragma unroll
    for (int j = 0; j < 4; ++j) {
      int idx = tid + j * THREADS;
      int row = idx >> 3, col = (idx & 7) * 8;
      stage_b128(src + idx * 8, dstbase + row * KLA + col);
    }
  };
  // stage one hidden layer: 1792 b128 xfers = 7/thread exact (16 xfers per 128-elem row)
  auto issue_wH = [&](int l) {
    const __bf16* src = wH + (size_t)l * (HP * KP);
    __bf16* dstbase = &wbufH[l & 1][0][0];
#pragma unroll
    for (int j = 0; j < 7; ++j) {
      int idx = tid + j * THREADS;
      int row = idx >> 4, col = (idx & 15) * 8;
      stage_b128(src + idx * 8, dstbase + row * KLH + col);
    }
  };

  issue_wH(0);   // hidden layer 0 streams behind the whole stage-0 GEMM
  issue_wA(0);

  // ===== Stage 0: h = relu(x @ w_in + b_in), K = 3072 =====
  v8f acc[NT];
#pragma unroll
  for (int t = 0; t < NT; ++t) acc[t] = {};

  const float* xrow = x + (size_t)(rowBase + laneLo) * D_IN;
  for (int c = 0; c < NCH; ++c) {
    if (c + 1 < NCH) { issue_wA(c + 1); WAIT_ASYNC(4); }  // retire chunk c (in-order)
    else             { WAIT_ASYNC(0); }
    __syncthreads();                                      // chunk c visible to all

    const __bf16* wb = &wbufA[c & 1][0][0];
    const int k0 = c * KCH;
#pragma unroll
    for (int j = 0; j < 2; ++j) {
      const int kb = k0 + j * 32 + kOffA;
      float4 f0 = *(const float4*)(xrow + kb);
      float4 f1 = *(const float4*)(xrow + kb + 4);
      float4 f2 = *(const float4*)(xrow + kb + 16);
      float4 f3 = *(const float4*)(xrow + kb + 20);
      __builtin_prefetch(xrow + kb + 1024, 0, 0);   // prefetch the HBM stream ahead
      v16bf a = cat16(cvt8(f0, f1), cvt8(f2, f3));
#pragma unroll
      for (int t = 0; t < NT; ++t) {
        const __bf16* bp = wb + (t * 16 + laneLo) * KLA + j * 32 + kOffB;
        acc[t] = wmma_bf16(a, ldB(bp), acc[t]);
      }
    }
    __syncthreads();   // all waves done with buffer (c&1) before chunk c+2 overwrites
  }

  // bias + relu -> h tile in LDS (bf16); zero the K-pad tail once
#pragma unroll
  for (int t = 0; t < NT; ++t) {
    int cc = t * 16 + laneLo;
    float bias = (cc < H) ? b_in[cc] : 0.0f;
#pragma unroll
    for (int r = 0; r < 8; ++r) {
      float v = acc[t][r] + bias;
      hbuf[wave][r + laneHi * 8][cc] = (__bf16)(v > 0.0f ? v : 0.0f);
    }
  }
#pragma unroll
  for (int r = 0; r < 8; ++r)
    hbuf[wave][r + laneHi * 8][HP + laneLo] = (__bf16)0.0f;

  // ===== per-wave helpers for the hidden chain =====
  v16bf af[4];
  auto load_af = [&]() {
#pragma unroll
    for (int kc = 0; kc < 4; ++kc) {
      const __bf16* hp = &hbuf[wave][laneLo][kc * 32 + kOffA];
      af[kc] = cat16(*(const v8bf*)hp, *(const v8bf*)(hp + 16));  // A layout: K, K+16
    }
  };

  auto hidden = [&](int l) {
    const __bf16* wl = &wbufH[l & 1][0][0];
    const float*  bl = b_hid + (size_t)l * H;
    v8f hacc[NT];
#pragma unroll
    for (int t = 0; t < NT; ++t) hacc[t] = {};
#pragma unroll
    for (int t = 0; t < NT; ++t) {
      const __bf16* wrow = wl + (t * 16 + laneLo) * KLH + kOffB;
#pragma unroll
      for (int kc = 0; kc < 4; ++kc)
        hacc[t] = wmma_bf16(af[kc], ldB(wrow + kc * 32), hacc[t]);
    }
#pragma unroll
    for (int t = 0; t < NT; ++t) {
      int cc = t * 16 + laneLo;
      float bias = (cc < H) ? bl[cc] : 0.0f;
#pragma unroll
      for (int r = 0; r < 8; ++r) {
        float v = hacc[t][r] + bias;
        hbuf[wave][r + laneHi * 8][cc] = (__bf16)(v > 0.0f ? v : 0.0f);
      }
    }
  };

  auto outstage = [&](int s) {
    const __bf16* wsrow = wO + (size_t)s * 16 * KP + (size_t)laneLo * KP + kOffB;
    v8f oa = {};
#pragma unroll
    for (int kc = 0; kc < 4; ++kc)
      oa = wmma_bf16(af[kc], ldB(wsrow + kc * 32), oa);
    float bias = (laneLo < OUT_N) ? b_out[s * OUT_N + laneLo] : 0.0f;
#pragma unroll
    for (int r = 0; r < 8; ++r)
      obuf[wave][s][r + laneHi * 8][laneLo] = oa[r] + bias;
  };

  // ===== hidden chain: 4 layers -> head0, 5 -> head1, 5 -> head2 =====
  load_af();
  for (int l = 0; l < 14; ++l) {
    if (l + 1 < 14) { issue_wH(l + 1); WAIT_ASYNC(7); }  // retire layer l's copies
    else            { WAIT_ASYNC(0); }
    __syncthreads();            // layer l weights visible
    hidden(l);                  // consume af (= h_l), write h_{l+1} to hbuf
    load_af();                  // af = h_{l+1}
    if      (l == 3)  outstage(0);
    else if (l == 8)  outstage(1);
    else if (l == 13) outstage(2);
    __syncthreads();            // buffer (l&1) free before layer l+2 streams into it
  }

  __syncthreads();

  // ===== early-exit select (dense-masked equivalent) =====
  if (lane < 16) {
    int row = lane;
    const float* o0 = &obuf[wave][0][row][0];
    const float* o1 = &obuf[wave][1][row][0];
    const float* o2 = &obuf[wave][2][row][0];
    float m0 = o0[0], m1 = o1[0];
#pragma unroll
    for (int c = 1; c < OUT_N; ++c) {
      m0 = fmaxf(m0, o0[c]);
      m1 = fmaxf(m1, o1[c]);
    }
    const float* src = (m0 > CUTOFF) ? o0 : ((m1 > CUTOFF) ? o1 : o2);
    float* dst = out + (size_t)(rowBase + row) * OUT_N;
#pragma unroll
    for (int c = 0; c < OUT_N; ++c) dst[c] = src[c];
  }
}

// ---------------- launch ----------------

extern "C" void kernel_launch(void* const* d_in, const int* in_sizes, int n_in,
                              void* d_out, int out_size, void* d_ws, size_t ws_size,
                              hipStream_t stream) {
  const float* x     = (const float*)d_in[0];
  const float* w_in  = (const float*)d_in[1];
  const float* b_in  = (const float*)d_in[2];
  const float* w_hid = (const float*)d_in[3];
  const float* b_hid = (const float*)d_in[4];
  const float* w_out = (const float*)d_in[5];
  const float* b_out = (const float*)d_in[6];
  float* out = (float*)d_out;

  __bf16* wA = (__bf16*)d_ws;
  __bf16* wH = wA + WA_ELEMS;
  __bf16* wO = wH + WH_ELEMS;

  prep_wa<<<(WA_ELEMS + 255) / 256, 256, 0, stream>>>(w_in,  wA);
  prep_wh<<<(WH_ELEMS + 255) / 256, 256, 0, stream>>>(w_hid, wH);
  prep_wo<<<(WO_ELEMS + 255) / 256, 256, 0, stream>>>(w_out, wO);

  dim3 grid(NROWS / ROWS_PER_BLOCK);   // 512 blocks
  dim3 block(THREADS);                 // 8 waves (wave32)
  mlp_cascade_fused<<<grid, block, 0, stream>>>(x, b_in, b_hid, b_out,
                                                wA, wH, wO, out);
}
